// LTIModel_45449343927016
// MI455X (gfx1250) — matrix operand order repaired and checked by reading, compiled, not run on hardware
//
#include <hip/hip_runtime.h>
#include <hip/hip_bf16.h>

// ---------------- problem constants (from reference) ----------------
#define BTOT   2048
#define TSTEPS 256
#define SDIM   13
#define CDIM   4
#define LDIM   256
#define ROWS   16          // rows of the batch handled per workgroup
#define ZPAD   20          // padded row stride of transposed f32 state (bank-friendly)
#define DT_H   0.01f

typedef __attribute__((ext_vector_type(16))) __bf16 v16bf;
typedef __attribute__((ext_vector_type(8)))  float  v8f;
typedef unsigned int u32x4 __attribute__((ext_vector_type(4)));
typedef float        f32x4 __attribute__((ext_vector_type(4)));

union BF32 { v16bf v; u32x4 q[2]; unsigned short s[16]; };

__device__ inline unsigned short f2bf(float f) {
  unsigned int u = __builtin_bit_cast(unsigned int, f);
  unsigned int r = u + 0x7FFFu + ((u >> 16) & 1u);   // round-to-nearest-even
  return (unsigned short)(r >> 16);
}

// ---------------- kernel 1: build M = I - (dt/2)A, N = I + (dt/2)A ----------------
__global__ void lti_build_mn(const float* __restrict__ A_ct,
                             float* __restrict__ Mw, float* __restrict__ Nw) {
  int i = blockIdx.x, j = threadIdx.x;
  float f = 0.5f * DT_H * A_ct[i * LDIM + j];
  float d = (i == j) ? 1.0f : 0.0f;
  Mw[i * LDIM + j] = d - f;
  Nw[i * LDIM + j] = d + f;
}

// ---------------- kernel 2: Gauss-Jordan: N <- M^{-1} N  (single workgroup) ----------------
__global__ __launch_bounds__(1024) void lti_gauss_jordan(float* __restrict__ Mw,
                                                         float* __restrict__ Nw) {
  const int tid = threadIdx.x;
  __shared__ float pinv;
  for (int p = 0; p < LDIM; ++p) {
    if (tid == 0) pinv = 1.0f / Mw[p * LDIM + p];
    __syncthreads();
    float pv = pinv;
    if (tid < LDIM)            Mw[p * LDIM + tid] *= pv;
    else if (tid < 2 * LDIM)   Nw[p * LDIM + (tid - LDIM)] *= pv;
    __syncthreads();
    const int r  = tid >> 2;          // 0..255
    const int j0 = (tid & 3) * 64;    // 64-col slice
    float fac = Mw[r * LDIM + p];     // read BEFORE anyone zeroes column p
    __syncthreads();
    if (r != p) {
      for (int j = j0; j < j0 + 64; ++j) {
        Mw[r * LDIM + j] -= fac * Mw[p * LDIM + j];
        Nw[r * LDIM + j] -= fac * Nw[p * LDIM + j];
      }
    }
    __syncthreads();
  }
}

// ---------------- kernel 3: E_bf16 = bf16(A_dt - I) ----------------
__global__ void lti_make_e(const float* __restrict__ Nw, unsigned short* __restrict__ Ebf) {
  int i = blockIdx.x, j = threadIdx.x;
  float e = Nw[i * LDIM + j] - ((i == j) ? 1.0f : 0.0f);
  Ebf[i * LDIM + j] = f2bf(e);
}

// ---------------- kernel 4: fused encode + rollout + decode ----------------
// One workgroup = 256 threads = 8 waves, owns ROWS=16 batch rows for all T steps.
// Wave w owns output columns [32w, 32w+32) of z; E^T operand tiles live in VGPRs.
// f32 master state kept TRANSPOSED (zft[n][m], padded stride) -> b128 LDS accesses.
// The u @ B^T rank-4 update is fed through WMMA as an extra K-block (bf16 staged u).
__global__ __launch_bounds__(256) void lti_rollout(
    const float* __restrict__ x_init,      // [BTOT][SDIM]
    const float* __restrict__ u_future,    // [BTOT][TSTEPS][CDIM]
    const float* __restrict__ W_enc,       // [LDIM][SDIM]
    const float* __restrict__ W_dec,       // [SDIM][LDIM]
    const float* __restrict__ Bmat,        // [LDIM][CDIM]
    const unsigned short* __restrict__ Ebf,// [LDIM][LDIM] bf16, E[n][k]
    float* __restrict__ out)               // [BTOT][TSTEPS+1][SDIM]
{
  __shared__ __attribute__((aligned(16))) float          zft[2][LDIM][ZPAD];  // f32 state, transposed
  __shared__ __attribute__((aligned(16))) unsigned short zh[2][ROWS][LDIM];  // bf16 copy (A-operand)
  __shared__ __attribute__((aligned(16))) unsigned short ush[2][ROWS][32];   // u as bf16 A-tile (cols 4..31 = 0)
  __shared__ float xacc[2][ROWS][16];                                        // decode accumulators

  const int tid  = threadIdx.x;
  const int lane = tid & 31;
  const int wave = tid >> 5;          // 0..7
  const int half = lane >> 4;         // lane group
  const int nl   = lane & 15;
  const int rowbase = blockIdx.x * ROWS;

  // ---- per-wave register-resident B-operand tiles of E^T (loaded once) ----
  // B[k][n] = E[Nbase+n][k]; contiguous k order (same as A loads): lanes0-15 K=0..15, lanes16-31 K=16..31, packed pairs.
  BF32 btile[8][2];
#pragma unroll
  for (int kb = 0; kb < 8; ++kb)
#pragma unroll
    for (int tl = 0; tl < 2; ++tl) {
      int n = 32 * wave + 16 * tl + nl;
      const u32x4* p = (const u32x4*)(Ebf + n * LDIM + 32 * kb + 16 * half);
      btile[kb][tl].q[0] = p[0];
      btile[kb][tl].q[1] = p[1];
    }

  // ---- decode B-operand: B[k][n] = W_dec[n][k], wave's K chunk, n>=SDIM zero ----
  BF32 bdec;
#pragma unroll
  for (int j = 0; j < 8; ++j) {
    int k = 32 * wave + 16 * half + 2 * j;
    unsigned short lo = 0, hi = 0;
    if (nl < SDIM) { lo = f2bf(W_dec[nl * LDIM + k]); hi = f2bf(W_dec[nl * LDIM + k + 1]); }
    bdec.s[2 * j]     = lo;
    bdec.s[2 * j + 1] = hi;
  }

  // ---- u-term B-operand: B[k][n] = Bmat[n][k] for k<4, else 0 ----
  BF32 bu[2];
#pragma unroll
  for (int tl = 0; tl < 2; ++tl) {
    int n = 32 * wave + 16 * tl + nl;
#pragma unroll
    for (int j = 0; j < 16; ++j) bu[tl].s[j] = 0;
    if (half == 0) {
#pragma unroll
      for (int c = 0; c < CDIM; ++c) bu[tl].s[c] = f2bf(Bmat[n * CDIM + c]);
    }
  }

  // ---- encoder lift: z0 = x_init @ W_enc^T ----
  if (tid < ROWS * SDIM) {
    int r = tid / SDIM, s = tid % SDIM;
    xacc[0][r][s] = x_init[(rowbase + r) * SDIM + s];   // stage x_init
  }
  __syncthreads();
  {
    float we[SDIM];
#pragma unroll
    for (int s = 0; s < SDIM; ++s) we[s] = W_enc[tid * SDIM + s];
    for (int r = 0; r < ROWS; ++r) {
      float acc = 0.0f;
#pragma unroll
      for (int s = 0; s < SDIM; ++s) acc += xacc[0][r][s] * we[s];
      zft[0][tid][r] = acc;           // transposed store: column tid, row r
      zh[0][r][tid]  = f2bf(acc);
    }
    // zero u staging tiles (cols 4..31 stay zero forever)
    unsigned int* uz = (unsigned int*)ush;
    uz[tid] = 0u;
    uz[tid + 256] = 0u;
  }
  __syncthreads();
  { int r = tid >> 4, cc = tid & 15; xacc[0][r][cc] = 0.0f; xacc[1][r][cc] = 0.0f; }
  if (tid < ROWS) {
    const float* up = u_future + (size_t)(rowbase + tid) * TSTEPS * CDIM;
#pragma unroll
    for (int c = 0; c < CDIM; ++c) ush[0][tid][c] = f2bf(up[c]);
  }
  __syncthreads();

  const int n0 = 32 * wave + nl;      // tile-0 output column for this lane
  const int n1 = n0 + 16;             // tile-1 output column

  // ---- main rollout: iteration t decodes z_t and (t<T) computes z_{t+1} ----
  for (int t = 0; t <= TSTEPS; ++t) {
    const int cur = t & 1, nxt = cur ^ 1;

    // (1) decode partial: xacc[cur] += zh[cur](K-chunk of this wave) x W_dec^T
    {
      BF32 a;
      a.q[0] = *(const u32x4*)(&zh[cur][nl][32 * wave + 16 * half]);
      a.q[1] = *(const u32x4*)(&zh[cur][nl][32 * wave + 16 * half + 8]);
      v8f cz = {};
      v8f d = __builtin_amdgcn_wmma_f32_16x16x32_bf16(
          false, a.v, false, bdec.v, (short)0, cz, false, false);
#pragma unroll
      for (int i = 0; i < 8; ++i)
        atomicAdd(&xacc[cur][i + 8 * half][nl], d[i]);   // ds_add_f32
    }

    if (t < TSTEPS) {
      // (2) C-init: pure f32 state read (transposed b128 loads)
      const f32x4* zp0 = (const f32x4*)&zft[cur][n0][8 * half];
      const f32x4* zp1 = (const f32x4*)&zft[cur][n1][8 * half];
      f32x4 za = zp0[0], zb = zp0[1];
      f32x4 zc = zp1[0], zd = zp1[1];
      v8f c0, c1;
#pragma unroll
      for (int i = 0; i < 8; ++i) {
        c0[i] = (i < 4) ? za[i & 3] : zb[i & 3];
        c1[i] = (i < 4) ? zc[i & 3] : zd[i & 3];
      }

      // rank-4 u-term via WMMA (u staged as bf16 A-tile)
      {
        BF32 au;
        au.q[0] = *(const u32x4*)(&ush[cur][nl][8 * half]);
        au.q[1] = *(const u32x4*)(&ush[cur][nl][16 + 8 * half]);
        c0 = __builtin_amdgcn_wmma_f32_16x16x32_bf16(
            false, au.v, false, bu[0].v, (short)0, c0, false, false);
        c1 = __builtin_amdgcn_wmma_f32_16x16x32_bf16(
            false, au.v, false, bu[1].v, (short)0, c1, false, false);
      }

      // residual GEMM: load each A-tile once, feed both N-tiles
#pragma unroll
      for (int kb = 0; kb < 8; ++kb) {
        BF32 a;
        a.q[0] = *(const u32x4*)(&zh[cur][nl][32 * kb + 16 * half]);
        a.q[1] = *(const u32x4*)(&zh[cur][nl][32 * kb + 16 * half + 8]);
        c0 = __builtin_amdgcn_wmma_f32_16x16x32_bf16(
            false, a.v, false, btile[kb][0].v, (short)0, c0, false, false);
        c1 = __builtin_amdgcn_wmma_f32_16x16x32_bf16(
            false, a.v, false, btile[kb][1].v, (short)0, c1, false, false);
      }

      // write-back: f32 state as b128 stores (transposed), bf16 copy row-major
      {
        f32x4 w;
        w = (f32x4){c0[0], c0[1], c0[2], c0[3]};
        ((f32x4*)&zft[nxt][n0][8 * half])[0] = w;
        w = (f32x4){c0[4], c0[5], c0[6], c0[7]};
        ((f32x4*)&zft[nxt][n0][8 * half])[1] = w;
        w = (f32x4){c1[0], c1[1], c1[2], c1[3]};
        ((f32x4*)&zft[nxt][n1][8 * half])[0] = w;
        w = (f32x4){c1[4], c1[5], c1[6], c1[7]};
        ((f32x4*)&zft[nxt][n1][8 * half])[1] = w;
      }
#pragma unroll
      for (int i = 0; i < 8; ++i) {
        int m = i + 8 * half;
        zh[nxt][m][n0] = f2bf(c0[i]);
        zh[nxt][m][n1] = f2bf(c1[i]);
      }

      // (3) stage u[t+1] (bf16); prefetch u[t+2]
      if (t + 1 < TSTEPS && tid < ROWS) {
        const float* up = u_future + (size_t)(rowbase + tid) * TSTEPS * CDIM
                        + (size_t)(t + 1) * CDIM;
#pragma unroll
        for (int c = 0; c < CDIM; ++c) ush[nxt][tid][c] = f2bf(up[c]);
        __builtin_prefetch(up + CDIM, 0, 3);   // global_prefetch_b8
      }
    }

    // (4) flush x(t-1) held in xacc[nxt], re-zero it (same thread reads+zeroes)
    if (t >= 1 && tid < ROWS * SDIM) {
      int r = tid / SDIM, s = tid % SDIM;
      out[(size_t)(rowbase + r) * (TSTEPS + 1) * SDIM + (size_t)(t - 1) * SDIM + s]
          = xacc[nxt][r][s];
      xacc[nxt][r][s] = 0.0f;
    }
    __syncthreads();
  }

  // final flush: x(T)
  if (tid < ROWS * SDIM) {
    int r = tid / SDIM, s = tid % SDIM;
    out[(size_t)(rowbase + r) * (TSTEPS + 1) * SDIM + (size_t)TSTEPS * SDIM + s]
        = xacc[TSTEPS & 1][r][s];
  }
}

// ---------------- launcher ----------------
extern "C" void kernel_launch(void* const* d_in, const int* in_sizes, int n_in,
                              void* d_out, int out_size, void* d_ws, size_t ws_size,
                              hipStream_t stream) {
  (void)in_sizes; (void)n_in; (void)out_size; (void)ws_size;
  // setup_inputs order:
  // 0:x_history 1:u_history 2:x_init 3:x_future 4:u_future 5:A_ct 6:B 7:W_enc 8:W_dec 9:n_steps
  const float* x_init   = (const float*)d_in[2];
  const float* u_future = (const float*)d_in[4];
  const float* A_ct     = (const float*)d_in[5];
  const float* Bmat     = (const float*)d_in[6];
  const float* W_enc    = (const float*)d_in[7];
  const float* W_dec    = (const float*)d_in[8];
  float*       out      = (float*)d_out;

  // workspace layout: M f32 (256KB) | N f32 (256KB) | E bf16 (128KB)
  float*          Mw  = (float*)d_ws;
  float*          Nw  = Mw + LDIM * LDIM;
  unsigned short* Ebf = (unsigned short*)((char*)d_ws + 2u * LDIM * LDIM * sizeof(float));

  lti_build_mn    <<<dim3(LDIM), dim3(LDIM), 0, stream>>>(A_ct, Mw, Nw);
  lti_gauss_jordan<<<dim3(1),    dim3(1024), 0, stream>>>(Mw, Nw);
  lti_make_e      <<<dim3(LDIM), dim3(LDIM), 0, stream>>>(Nw, Ebf);
  lti_rollout     <<<dim3(BTOT / ROWS), dim3(256), 0, stream>>>(
      x_init, u_future, W_enc, W_dec, Bmat, Ebf, out);
}
